// GemmaAttention_38457137169042
// MI455X (gfx1250) — matrix operand order repaired and checked by reading, not compiled, not run on hardware
//
#include <hip/hip_runtime.h>
#include <hip/hip_bf16.h>
#include <stdint.h>

// ---------------------------------------------------------------------------
// CDNA5 (gfx1250) bf16 WMMA implementation of Gemma attention block.
// - All GEMMs + attention matmuls via v_wmma_f32_16x16x32_bf16 (f32 accum).
// - GEMM B-tiles staged through LDS by the Tensor Data Mover (TDM),
//   double-buffered, synced with s_wait_tensorcnt + workgroup barrier.
// ---------------------------------------------------------------------------

typedef __attribute__((ext_vector_type(16))) __bf16 v16bf;
typedef __attribute__((ext_vector_type(8)))  __bf16 v8bf;
typedef __attribute__((ext_vector_type(8)))  float  v8f;
typedef __attribute__((ext_vector_type(4)))  unsigned int u32x4;
typedef __attribute__((ext_vector_type(8)))  int i32x8;
typedef __attribute__((ext_vector_type(4)))  int i32x4;

#define WMMA_BF16(a, b, c) \
  __builtin_amdgcn_wmma_f32_16x16x32_bf16(false, (a), false, (b), (short)0, (c), false, false)

#define NEG_MASK (-2.3819763e+38f)

// ------------------------- fp32 -> bf16 conversion -------------------------
__global__ void k_cvt_bf16(const float* __restrict__ src,
                           __bf16* __restrict__ dst, long long n) {
  long long i = (long long)blockIdx.x * blockDim.x + threadIdx.x;
  if (i < n) dst[i] = (__bf16)src[i];
}

// ---------------- TDM: stage a 64(N) x 32(K) bf16 tile into LDS ------------
// D# per CDNA5 ISA ch.8: group0 = {count/flags, lds_addr, global_addr, type},
// group1 = {mask/data_size, tensor dims, tile dims, strides}.  2D tile:
// dim0 (contiguous) = K (32 elements/row), dim1 = N (64 rows).
__device__ __forceinline__ void tdm_load_b_tile(const __bf16* gsrc,
                                                unsigned lds_off,
                                                int K, int N) {
  const unsigned long long ga = (unsigned long long)(uintptr_t)gsrc;
  u32x4 g0;
  g0[0] = 1u;                                        // count=1, user mode
  g0[1] = lds_off;                                   // lds_addr (bytes)
  g0[2] = (unsigned)(ga & 0xffffffffull);            // global_addr[31:0]
  g0[3] = (unsigned)((ga >> 32) & 0x1ffffffull) | (2u << 30);  // addr hi, type=2

  const unsigned uk = (unsigned)K, un = (unsigned)N;
  i32x8 g1;
  g1[0] = (int)(1u << 16);                           // data_size=1 (2 bytes)
  g1[1] = (int)((uk & 0xffffu) << 16);               // tensor_dim0[15:0]
  g1[2] = (int)(((uk >> 16) & 0xffffu) | ((un & 0xffffu) << 16));  // d0 hi|d1 lo
  g1[3] = (int)(((un >> 16) & 0xffffu) | (32u << 16));             // d1 hi|tile0=32
  g1[4] = (int)64u;                                  // tile_dim1=64, tile_dim2=0
  g1[5] = (int)uk;                                   // tensor_dim0_stride lo32
  g1[6] = 0;                                         // stride0 hi, stride1 lo
  g1[7] = 0;
  i32x4 z4 = {0, 0, 0, 0};
  i32x8 z8 = {0, 0, 0, 0, 0, 0, 0, 0};
  __builtin_amdgcn_tensor_load_to_lds(g0, g1, z4, z4, z8, 0);
}

// ----------------------------- WMMA GEMM -----------------------------------
// C[M,N] = A[M,K] * Bw[N,K]^T   (A, Bw row-major bf16)
// MODE 0: bf16 row-major out; MODE 1: f32 row-major out;
// MODE 2: bf16 out stored as VT[b][n][s] (assumes M = b*2048 + s, N = 2048).
template <int MODE>
__global__ __launch_bounds__(256)
void k_wmma_gemm(const __bf16* __restrict__ A, const __bf16* __restrict__ Bw,
                 void* __restrict__ Cout, int M, int N, int K) {
  __shared__ __attribute__((aligned(128))) __bf16 bstage[2][64 * 32];
  const int lane = threadIdx.x & 31;
  const int wave = threadIdx.x >> 5;
  const int h16  = lane >> 4;
  const int l16  = lane & 15;
  const int m0 = blockIdx.x * 128 + (wave >> 1) * 32;
  const int n0wg = blockIdx.y * 64;               // workgroup N origin
  const int n0l  = (wave & 1) * 32;               // wave N offset inside stage
  (void)M;

  const unsigned lds0 = (unsigned)(uintptr_t)&bstage[0][0];
  const unsigned lds1 = (unsigned)(uintptr_t)&bstage[1][0];

  // prologue: TDM-stage first B tile (wave 0 only; TDM is per-wave)
  if (threadIdx.x < 32) {
    tdm_load_b_tile(Bw + (size_t)n0wg * K, lds0, K, N);
    __builtin_amdgcn_s_wait_tensorcnt(0);
  }
  __syncthreads();

  v8f acc[2][2] = {};
  int cur = 0;
  for (int k0 = 0; k0 < K; k0 += 32, cur ^= 1) {
    // kick off next B tile into the other buffer (overlaps with WMMA below)
    if (k0 + 32 < K && threadIdx.x < 32)
      tdm_load_b_tile(Bw + (size_t)n0wg * K + (k0 + 32), cur ? lds0 : lds1, K, N);

    // A fragments (two 16-row tiles) straight from global (L2/WGP$ resident)
    v16bf af[2];
#pragma unroll
    for (int i = 0; i < 2; ++i) {
      const __bf16* p = A + (size_t)(m0 + 16 * i + l16) * K + k0 + 16 * h16;
      v8bf lo = *(const v8bf*)p;
      v8bf hi = *(const v8bf*)(p + 8);
#pragma unroll
      for (int e = 0; e < 8; ++e) { af[i][e] = lo[e]; af[i][e + 8] = hi[e]; }
    }
    if (k0 + 64 < K)
      __builtin_prefetch(A + (size_t)(m0 + l16) * K + k0 + 64, 0, 3);

    // B fragments from the staged LDS tile: row nl holds 32 contiguous K vals
    v16bf bfr[2];
#pragma unroll
    for (int j = 0; j < 2; ++j) {
      const __bf16* p = &bstage[cur][(n0l + 16 * j + l16) * 32 + 16 * h16];
      bfr[j] = *(const v16bf*)p;
    }

#pragma unroll
    for (int i = 0; i < 2; ++i)
#pragma unroll
      for (int j = 0; j < 2; ++j)
        acc[i][j] = WMMA_BF16(af[i], bfr[j], acc[i][j]);

    if (threadIdx.x < 32) __builtin_amdgcn_s_wait_tensorcnt(0);
    __syncthreads();
  }

#pragma unroll
  for (int i = 0; i < 2; ++i)
#pragma unroll
    for (int j = 0; j < 2; ++j)
#pragma unroll
      for (int r = 0; r < 8; ++r) {
        const int row = m0 + 16 * i + r + 8 * h16;
        const int col = n0wg + n0l + 16 * j + l16;
        const float v = acc[i][j][r];
        if (MODE == 1) {
          ((float*)Cout)[(size_t)row * N + col] = v;
        } else if (MODE == 2) {
          // VT[b][n][s]: b = row/2048, s = row%2048
          ((__bf16*)Cout)[((size_t)(row >> 11) * 2048 + col) * 2048 + (row & 2047)] =
              (__bf16)v;
        } else {
          ((__bf16*)Cout)[(size_t)row * N + col] = (__bf16)v;
        }
      }
}

// ------------------------------- RoPE --------------------------------------
__global__ void k_rope(__bf16* __restrict__ x, const float* __restrict__ cs,
                       const float* __restrict__ sn, int heads, float scale) {
  const int per = heads * 128;
  const int idx = blockIdx.x * blockDim.x + threadIdx.x;
  if (idx >= 8192 * per) return;
  const int row = idx / per;
  const int rem = idx - row * per;
  const int h = rem >> 7;
  const int d = rem & 127;
  const float c = cs[row * 128 + d];
  const float s = sn[row * 128 + d];
  __bf16* p = x + (size_t)row * (heads * 256) + h * 256 + d;
  const float x1 = (float)p[0];
  const float x2 = (float)p[128];
  p[0]   = (__bf16)(scale * (x1 * c - x2 * s));
  p[128] = (__bf16)(scale * (x1 * s + x2 * c));
}

// --------------------------- flash attention -------------------------------
__device__ __forceinline__ float redmax16(float v) {
#pragma unroll
  for (int m = 1; m < 16; m <<= 1) v = fmaxf(v, __shfl_xor(v, m, 32));
  return v;
}
__device__ __forceinline__ float redsum16(float v) {
#pragma unroll
  for (int m = 1; m < 16; m <<= 1) v += __shfl_xor(v, m, 32);
  return v;
}

// grid = (S/64, NQ, B), block = 128 (4 waves, each owns 16 query rows)
__global__ __launch_bounds__(128)
void k_flash(const __bf16* __restrict__ Q, const __bf16* __restrict__ Kc,
             const __bf16* __restrict__ VT, __bf16* __restrict__ AO) {
  __shared__ __bf16 plds[4 * 512];  // per-wave 16x32 bf16 P tile
  const int lane = threadIdx.x & 31;
  const int wave = threadIdx.x >> 5;
  const int h16 = lane >> 4;
  const int l16 = lane & 15;
  const int b = blockIdx.z;
  const int h = blockIdx.y;
  const int kvh = h >> 1;            // GQA group of 2
  const int q0 = blockIdx.x * 64 + wave * 16;

  const __bf16* Qb = Q  + (size_t)b * 2048 * 4096 + h * 256;
  const __bf16* Kb = Kc + (size_t)b * 2048 * 2048 + kvh * 256;
  const __bf16* Vb = VT + (size_t)(b * 8 + kvh) * 256 * 2048;  // [d][s]
  __bf16* lbuf = plds + wave * 512;

  v8f acc[16] = {};                  // O tile: 16 rows x 256 d (f32)
  float mrow[8], lrow[8];
#pragma unroll
  for (int r = 0; r < 8; ++r) { mrow[r] = -3.0e38f; lrow[r] = 0.f; }

  for (int sk0 = 0; sk0 < q0 + 16; sk0 += 32) {
    // ---- scores S[16 x 32] = Q(16x256) . K^T : two C tiles over D ----
    v8f s0 = {}, s1 = {};
#pragma unroll
    for (int d0 = 0; d0 < 256; d0 += 32) {
      v16bf qa;
      {
        const __bf16* p = Qb + (size_t)(q0 + l16) * 4096 + d0 + 16 * h16;
        v8bf lo = *(const v8bf*)p;
        v8bf hi = *(const v8bf*)(p + 8);
#pragma unroll
        for (int e = 0; e < 8; ++e) { qa[e] = lo[e]; qa[e + 8] = hi[e]; }
      }
      v16bf b0 = *(const v16bf*)(Kb + (size_t)(sk0 + l16) * 2048 + d0 + 16 * h16);
      v16bf b1 = *(const v16bf*)(Kb + (size_t)(sk0 + 16 + l16) * 2048 + d0 + 16 * h16);
      s0 = WMMA_BF16(qa, b0, s0);
      s1 = WMMA_BF16(qa, b1, s1);
    }

    // ---- causal mask on the diagonal blocks ----
    if (sk0 + 31 > q0) {
#pragma unroll
      for (int r = 0; r < 8; ++r) {
        const int row = q0 + r + 8 * h16;
        if (sk0 + l16 > row)      s0[r] = NEG_MASK;
        if (sk0 + 16 + l16 > row) s1[r] = NEG_MASK;
      }
    }

    // ---- online softmax update ----
#pragma unroll
    for (int r = 0; r < 8; ++r) {
      float mb = redmax16(fmaxf(s0[r], s1[r]));
      const float mn = fmaxf(mrow[r], mb);
      const float al = __expf(mrow[r] - mn);
      mrow[r] = mn;
      const float p0 = __expf(s0[r] - mn);
      const float p1 = __expf(s1[r] - mn);
      lrow[r] = lrow[r] * al + redsum16(p0 + p1);
#pragma unroll
      for (int t = 0; t < 16; ++t) acc[t][r] *= al;
      // park P in LDS (row-major 16x32) for C->A fragment re-layout
      lbuf[(r + 8 * h16) * 32 + l16]      = (__bf16)p0;
      lbuf[(r + 8 * h16) * 32 + 16 + l16] = (__bf16)p1;
    }
    __builtin_amdgcn_wave_barrier();  // keep ds_store ahead of ds_load

    // ---- rebuild P as an A-operand fragment (16x32) ----
    v16bf pa;
    {
      const __bf16* p = lbuf + l16 * 32 + 16 * h16;
      v8bf lo = *(const v8bf*)p;
      v8bf hi = *(const v8bf*)(p + 8);
#pragma unroll
      for (int e = 0; e < 8; ++e) { pa[e] = lo[e]; pa[e + 8] = hi[e]; }
    }

    // ---- O += P . V : 16 d-tiles, V^T rows contiguous over tokens ----
#pragma unroll
    for (int t = 0; t < 16; ++t) {
      v16bf vb =
          *(const v16bf*)(Vb + (size_t)(t * 16 + l16) * 2048 + sk0 + 16 * h16);
      acc[t] = WMMA_BF16(pa, vb, acc[t]);
    }
  }

  // ---- normalize and store attention output (bf16, row-major) ----
  __bf16* Ob = AO + (size_t)b * 2048 * 4096 + h * 256;
#pragma unroll
  for (int r = 0; r < 8; ++r) {
    const float inv = 1.f / lrow[r];
    const size_t rowoff = (size_t)(q0 + r + 8 * h16) * 4096;
#pragma unroll
    for (int t = 0; t < 16; ++t)
      Ob[rowoff + t * 16 + l16] = (__bf16)(acc[t][r] * inv);
  }
}

// ---------------------------------------------------------------------------
extern "C" void kernel_launch(void* const* d_in, const int* in_sizes, int n_in,
                              void* d_out, int out_size, void* d_ws, size_t ws_size,
                              hipStream_t stream) {
  (void)in_sizes; (void)n_in; (void)out_size; (void)ws_size;
  const float* hs = (const float*)d_in[0];
  const float* fc = (const float*)d_in[1];
  const float* fs = (const float*)d_in[2];
  /* d_in[3] = mask: causality handled analytically */
  const float* wq = (const float*)d_in[4];
  const float* wk = (const float*)d_in[5];
  const float* wv = (const float*)d_in[6];
  const float* wo = (const float*)d_in[7];

  char* ws = (char*)d_ws;
  size_t off = 0;
  auto alloc = [&](long long elems) -> __bf16* {
    __bf16* p = (__bf16*)(ws + off);
    off += (size_t)elems * 2;
    return p;
  };
  __bf16* Xb  = alloc(8192LL * 3072);     // hidden, bf16
  __bf16* Wqb = alloc(4096LL * 3072);
  __bf16* Wkb = alloc(2048LL * 3072);
  __bf16* Wvb = alloc(2048LL * 3072);
  __bf16* Wob = alloc(3072LL * 4096);
  __bf16* Qb  = alloc(8192LL * 4096);     // Q (post-RoPE, pre-scaled)
  __bf16* Kb  = alloc(8192LL * 2048);     // K (post-RoPE)
  __bf16* VTb = alloc(4LL * 2048 * 2048); // V transposed [b][n][s]
  __bf16* AOb = alloc(8192LL * 4096);     // attention output

  auto cvt = [&](const float* s, __bf16* d, long long n) {
    k_cvt_bf16<<<dim3((unsigned)((n + 255) / 256)), dim3(256), 0, stream>>>(s, d, n);
  };
  cvt(hs, Xb, 8192LL * 3072);
  cvt(wq, Wqb, 4096LL * 3072);
  cvt(wk, Wkb, 2048LL * 3072);
  cvt(wv, Wvb, 2048LL * 3072);
  cvt(wo, Wob, 3072LL * 4096);

  // QKV projections (WMMA bf16 GEMM, A @ W^T); V stored directly transposed
  k_wmma_gemm<0><<<dim3(8192 / 128, 4096 / 64), dim3(256), 0, stream>>>(
      Xb, Wqb, (void*)Qb, 8192, 4096, 3072);
  k_wmma_gemm<0><<<dim3(8192 / 128, 2048 / 64), dim3(256), 0, stream>>>(
      Xb, Wkb, (void*)Kb, 8192, 2048, 3072);
  k_wmma_gemm<2><<<dim3(8192 / 128, 2048 / 64), dim3(256), 0, stream>>>(
      Xb, Wvb, (void*)VTb, 8192, 2048, 3072);

  // RoPE (Q gets the 1/sqrt(256) = 1/16 score scale folded in)
  {
    int total = 8192 * 16 * 128;
    k_rope<<<dim3((total + 255) / 256), dim3(256), 0, stream>>>(Qb, fc, fs, 16, 0.0625f);
  }
  {
    int total = 8192 * 8 * 128;
    k_rope<<<dim3((total + 255) / 256), dim3(256), 0, stream>>>(Kb, fc, fs, 8, 1.0f);
  }

  // Flash attention: grid (S/64, NQ, B), 4 waves per block
  k_flash<<<dim3(32, 16, 4), dim3(128), 0, stream>>>(Qb, Kb, VTb, AOb);

  // Output projection -> f32 d_out
  k_wmma_gemm<1><<<dim3(8192 / 128, 3072 / 64), dim3(256), 0, stream>>>(
      AOb, Wob, d_out, 8192, 3072, 4096);
}
